// ToMe16_mlp_hd64_9732395892978
// MI455X (gfx1250) — matrix operand order repaired and checked by reading, not compiled, not run on hardware
//
#include <hip/hip_runtime.h>
#include <math.h>

// ---------------------------------------------------------------------------
// ToMe (6 static merge rounds) + MLP for MI455X / gfx1250, wave32, WMMA f16.
// - Score GEMM (a @ bb^T, K=64) fused with row argmax (no NxN score matrix in
//   HBM: saves ~340MB of traffic @ 23.3TB/s), software-pipelined B prefetch.
// - MLP GEMMs: 16x64 output strip per wave (4 accumulators, 4x A reuse,
//   4 independent v_wmma per K-step for load/compute overlap).
// - Stable descending argsort per batch via bitonic sort in LDS (<=32KB).
// ---------------------------------------------------------------------------

typedef _Float16 v16h __attribute__((ext_vector_type(16)));
typedef _Float16 v8h  __attribute__((ext_vector_type(8)));
typedef float    v8f  __attribute__((ext_vector_type(8)));

#define S_MAX 2304          // max HP (= p/2) across rounds; row-array stride
#define NBATCH 8

// Pack (score, col) so that u64 max == (max score, then smallest col).
__device__ __forceinline__ unsigned long long packScore(float s, unsigned j) {
  unsigned u = __float_as_uint(s);
  u = (u & 0x80000000u) ? ~u : (u | 0x80000000u);   // order-preserving f32->u32
  return ((unsigned long long)u << 32) | (unsigned)(~j);
}

__device__ __forceinline__ unsigned long long shflXor64(unsigned long long v, int m) {
  unsigned lo = __shfl_xor((unsigned)v, m, 32);
  unsigned hi = __shfl_xor((unsigned)(v >> 32), m, 32);
  return ((unsigned long long)hi << 32) | lo;
}

__device__ __forceinline__ v16h packA(const _Float16* arow, int off) {
  v16h Am;
  v8h alo = *(const v8h*)(arow + off);        // A: 16 contiguous K per lane
  v8h ahi = *(const v8h*)(arow + off + 8);
  #pragma unroll
  for (int q = 0; q < 8; ++q) { Am[q] = alo[q]; Am[q + 8] = ahi[q]; }
  return Am;
}

// ---- metric: mean over 16 heads of 64 dims, L2-normalize, f16, split even/odd
__global__ void metric_kernel(const float* __restrict__ x,
                              _Float16* __restrict__ a, _Float16* __restrict__ bb,
                              int p, int HP, int HPpad) {
  int lane = threadIdx.x & 31, wave = threadIdx.x >> 5;
  long item  = (long)blockIdx.x * 4 + wave;
  long items = (long)NBATCH * HPpad * 2;
  if (item >= items) return;
  int parity = (int)(item & 1);
  long rest  = item >> 1;
  int i = (int)(rest % HPpad);
  int b = (int)(rest / HPpad);
  _Float16* dst = (parity ? bb : a) + ((long)b * HPpad + i) * 64;
  if (i >= HP) {                       // zero pad rows (cannot influence argmax)
    dst[lane] = (_Float16)0.0f; dst[lane + 32] = (_Float16)0.0f; return;
  }
  int tok = 2 * i + parity;
  const float* xp = x + ((long)b * p + tok) * 1024;
  float m0 = 0.f, m1 = 0.f;
  #pragma unroll
  for (int h = 0; h < 16; ++h) { m0 += xp[h*64 + lane]; m1 += xp[h*64 + lane + 32]; }
  m0 *= (1.0f/16.0f); m1 *= (1.0f/16.0f);
  float ss = m0*m0 + m1*m1;
  #pragma unroll
  for (int m = 16; m >= 1; m >>= 1) ss += __shfl_xor(ss, m, 32);
  float inv = 1.0f / sqrtf(fmaxf(ss, 1e-30f));
  dst[lane]      = (_Float16)(m0 * inv);
  dst[lane + 32] = (_Float16)(m1 * inv);
}

__global__ void clear_u64(unsigned long long* __restrict__ p, int n) {
  int i = blockIdx.x * blockDim.x + threadIdx.x;
  if (i < n) p[i] = 0ull;
}

// ---- scores = a @ bb^T (16x16 tiles, K=64 as 2 x k32 WMMA) + fused argmax.
//      Next-tile B prefetch overlaps loads with WMMA + argmax VALU.
__global__ void scores_argmax_kernel(const _Float16* __restrict__ a,
                                     const _Float16* __restrict__ bb,
                                     unsigned long long* __restrict__ nb,
                                     int HP, int HPpad) {
  int lane = threadIdx.x & 31, wave = threadIdx.x >> 5;
  int it = blockIdx.x, b = blockIdx.y;
  int ntJ = HPpad >> 4;
  int i0  = it << 4;
  int l15 = lane & 15, lh = lane >> 4, kc = lh * 16;
  const _Float16* arow = a  + ((long)b * HPpad + i0 + l15) * 64;
  const _Float16* bbas = bb + ((long)b * HPpad + l15) * 64 + lh * 16;
  v16h A0 = packA(arow, kc);            // K = 0..31 slice (held in registers)
  v16h A1 = packA(arow, 32 + kc);       // K = 32..63 slice

  unsigned long long best[8];
  #pragma unroll
  for (int v = 0; v < 8; ++v) best[v] = 0ull;

  int jt = wave;                        // wave-uniform loop: EXEC all-1s at WMMA
  v16h Bc0 = {}, Bc1 = {};
  if (jt < ntJ) {
    const _Float16* br = bbas + (long)(jt << 4) * 64;
    Bc0 = *(const v16h*)(br);           // B: 16 contiguous K per lane
    Bc1 = *(const v16h*)(br + 32);
  }
  while (jt < ntJ) {
    int jn = jt + 4;
    v16h Bn0 = {}, Bn1 = {};
    if (jn < ntJ) {                     // prefetch next tile
      const _Float16* br = bbas + (long)(jn << 4) * 64;
      Bn0 = *(const v16h*)(br);
      Bn1 = *(const v16h*)(br + 32);
    }
    v8f c = {};
    c = __builtin_amdgcn_wmma_f32_16x16x32_f16(false, A0, false, Bc0, (short)0, c, false, false);
    c = __builtin_amdgcn_wmma_f32_16x16x32_f16(false, A1, false, Bc1, (short)0, c, false, false);
    int j = (jt << 4) + l15;
    if (j < HP) {                       // mask padded columns
      #pragma unroll
      for (int v = 0; v < 8; ++v) {
        unsigned long long pk = packScore(c[v], (unsigned)j);
        if (pk > best[v]) best[v] = pk;
      }
    }
    Bc0 = Bn0; Bc1 = Bn1;
    jt = jn;
  }
  #pragma unroll
  for (int v = 0; v < 8; ++v) {
    unsigned long long pb = best[v];
    for (int m = 1; m < 16; m <<= 1) {  // reduce across the 16-lane N group
      unsigned long long o = shflXor64(pb, m);
      if (o > pb) pb = o;
    }
    if (l15 == 0) {
      int row = i0 + v + lh * 8;        // D layout: M = v (+8 for hi half)
      if (row < HP)
        atomicMax(&nb[(long)b * S_MAX + row], pb);    // order-independent
    }
  }
}

// ---- per-batch stable descending argsort via bitonic sort in LDS ----
__global__ void sort_kernel(const unsigned long long* __restrict__ nb,
                            int* __restrict__ nodej, int* __restrict__ edge,
                            int HP, int n /*pow2 >= HP*/) {
  __shared__ unsigned long long sk[4096];             // 32KB of 320KB LDS
  int b = blockIdx.x, tid = threadIdx.x, nt = blockDim.x;
  for (int i = tid; i < n; i += nt) {
    if (i < HP) {
      unsigned long long pk = nb[(long)b * S_MAX + i];
      nodej[(long)b * S_MAX + i] = (int)(~(unsigned)pk);      // argmax column
      sk[i] = (pk & 0xFFFFFFFF00000000ull) | (unsigned)(~(unsigned)i);
    } else sk[i] = 0ull;                                      // sinks to the end
  }
  __syncthreads();
  for (unsigned k = 2; k <= (unsigned)n; k <<= 1)
    for (unsigned j = k >> 1; j > 0; j >>= 1) {
      for (unsigned i = tid; i < (unsigned)n; i += nt) {
        unsigned ixj = i ^ j;
        if (ixj > i) {
          unsigned long long xA = sk[i], xB = sk[ixj];
          bool up = ((i & k) == 0);
          if (up ? (xA < xB) : (xA > xB)) { sk[i] = xB; sk[ixj] = xA; }
        }
      }
      __syncthreads();
    }
  for (int t = tid; t < HP; t += nt)
    edge[(long)b * S_MAX + t] = (int)(~(unsigned)sk[t]);
}

// ---- merge: out = concat(unm, dst) of t = x*size; then scatter-add; then /size
__global__ void merge_copy_x(const float* __restrict__ xin, const float* __restrict__ szin,
                             const int* __restrict__ edge, float* __restrict__ xout,
                             int p, int r, int u, int pnew) {
  long idx = (long)blockIdx.x * blockDim.x + threadIdx.x;
  long total = (long)NBATCH * pnew * 1024;
  if (idx >= total) return;
  int c = (int)(idx & 1023);
  long t = idx >> 10;
  int pos = (int)(t % pnew), b = (int)(t / pnew);
  int tok = (pos < u) ? (2 * edge[(long)b * S_MAX + r + pos]) : (2 * (pos - u) + 1);
  float sz = szin ? szin[(long)b * p + tok] : 1.0f;
  xout[idx] = xin[((long)b * p + tok) * 1024 + c] * sz;
}

__global__ void merge_copy_size(const float* __restrict__ szin, const int* __restrict__ edge,
                                float* __restrict__ szout, int p, int r, int u, int pnew) {
  int idx = blockIdx.x * blockDim.x + threadIdx.x;
  int total = NBATCH * pnew;
  if (idx >= total) return;
  int pos = idx % pnew, b = idx / pnew;
  int tok = (pos < u) ? (2 * edge[(long)b * S_MAX + r + pos]) : (2 * (pos - u) + 1);
  szout[idx] = szin ? szin[(long)b * p + tok] : 1.0f;
}

__global__ void merge_scatter_x(const float* __restrict__ xin, const float* __restrict__ szin,
                                const int* __restrict__ edge, const int* __restrict__ nodej,
                                float* __restrict__ xout, int p, int r, int u, int pnew) {
  long idx = (long)blockIdx.x * blockDim.x + threadIdx.x;
  long total = (long)NBATCH * r * 1024;
  if (idx >= total) return;
  int c = (int)(idx & 1023);
  long t = idx >> 10;
  int s = (int)(t % r), b = (int)(t / r);
  int e = edge[(long)b * S_MAX + s];
  int tok = 2 * e;
  int d = nodej[(long)b * S_MAX + e];
  float sz = szin ? szin[(long)b * p + tok] : 1.0f;
  atomicAdd(&xout[((long)b * pnew + u + d) * 1024 + c],
            xin[((long)b * p + tok) * 1024 + c] * sz);
}

__global__ void merge_scatter_size(const float* __restrict__ szin, const int* __restrict__ edge,
                                   const int* __restrict__ nodej, float* __restrict__ szout,
                                   int p, int r, int u, int pnew) {
  int idx = blockIdx.x * blockDim.x + threadIdx.x;
  int total = NBATCH * r;
  if (idx >= total) return;
  int s = idx % r, b = idx / r;
  int e = edge[(long)b * S_MAX + s];
  int tok = 2 * e;
  int d = nodej[(long)b * S_MAX + e];
  float sz = szin ? szin[(long)b * p + tok] : 1.0f;
  atomicAdd(&szout[b * pnew + u + d], sz);
}

__global__ void merge_finalize(float* __restrict__ xout, const float* __restrict__ szout,
                               int pnew) {
  long idx = (long)blockIdx.x * blockDim.x + threadIdx.x;
  long total = (long)NBATCH * pnew * 1024;
  if (idx >= total) return;
  xout[idx] /= szout[idx >> 10];
}

// ---- f32 -> f16 conversions; weights transposed to N-major via LDS tiles ----
__global__ void convert_f16(const float* __restrict__ in, _Float16* __restrict__ out, long n) {
  long i = (long)blockIdx.x * blockDim.x + threadIdx.x;
  if (i < n) out[i] = (_Float16)in[i];
}

// w: K x N row-major (f32) -> wt: N x K row-major (f16). K, N multiples of 32.
__global__ void transpose_f16_tiled(const float* __restrict__ w, _Float16* __restrict__ wt,
                                    int K, int N) {
  __shared__ float tile[32][33];                      // +1 pad: no bank conflicts
  int nt = blockIdx.x * 32, kt = blockIdx.y * 32;
  #pragma unroll
  for (int q = 0; q < 32; q += 8) {
    int k = kt + threadIdx.y + q, n = nt + threadIdx.x;
    tile[threadIdx.y + q][threadIdx.x] = w[(long)k * N + n];   // coalesced read
  }
  __syncthreads();
  #pragma unroll
  for (int q = 0; q < 32; q += 8) {
    int n = nt + threadIdx.y + q, k = kt + threadIdx.x;
    wt[(long)n * K + k] = (_Float16)tile[threadIdx.x][threadIdx.y + q]; // coalesced write
  }
}

// ---- WMMA GEMM: 16x64 output strip per wave (4 accumulators, 4x A reuse) ----
template <bool GELU>
__global__ void gemm_wmma(const _Float16* __restrict__ A, const _Float16* __restrict__ Bt,
                          const float* __restrict__ bias, _Float16* __restrict__ outH,
                          float* __restrict__ outF, int M, int N, int K) {
  int lane = threadIdx.x & 31, wave = threadIdx.x >> 5;
  int ntN = N >> 6;                                   // 64-wide N strips
  int tiles = (M >> 4) * ntN;
  int tile = blockIdx.x * 8 + wave;
  if (tile >= tiles) return;                          // wave-uniform exit
  int mt = tile / ntN, nt_ = tile % ntN;
  int l15 = lane & 15, lh = lane >> 4, kc = lh * 16;
  const _Float16* arow = A  + (long)(mt * 16 + l15) * K;
  const _Float16* brow = Bt + (long)(nt_ * 64 + l15) * K + lh * 16;
  long bstep = (long)16 * K;                          // next 16-col B tile
  v8f c0 = {}, c1 = {}, c2 = {}, c3 = {};
  for (int k0 = 0; k0 < K; k0 += 32) {
    v16h Am = packA(arow, k0 + kc);                   // loaded once, used 4x
    v16h B0 = *(const v16h*)(brow + k0);
    v16h B1 = *(const v16h*)(brow + bstep + k0);
    v16h B2 = *(const v16h*)(brow + 2 * bstep + k0);
    v16h B3 = *(const v16h*)(brow + 3 * bstep + k0);
    c0 = __builtin_amdgcn_wmma_f32_16x16x32_f16(false, Am, false, B0, (short)0, c0, false, false);
    c1 = __builtin_amdgcn_wmma_f32_16x16x32_f16(false, Am, false, B1, (short)0, c1, false, false);
    c2 = __builtin_amdgcn_wmma_f32_16x16x32_f16(false, Am, false, B2, (short)0, c2, false, false);
    c3 = __builtin_amdgcn_wmma_f32_16x16x32_f16(false, Am, false, B3, (short)0, c3, false, false);
  }
  #pragma unroll
  for (int t = 0; t < 4; ++t) {
    v8f c = (t == 0) ? c0 : (t == 1) ? c1 : (t == 2) ? c2 : c3;
    int n = nt_ * 64 + t * 16 + l15;
    float bv = bias[n];
    #pragma unroll
    for (int v = 0; v < 8; ++v) {
      int row = mt * 16 + v + lh * 8;
      float val = c[v] + bv;
      if (GELU) {
        float g = 0.5f * val * (1.0f + erff(val * 0.70710678118654752f));
        outH[(long)row * N + n] = (_Float16)g;
      } else {
        outF[(long)row * N + n] = val;
      }
    }
  }
}

// ---------------------------------------------------------------------------
extern "C" void kernel_launch(void* const* d_in, const int* in_sizes, int n_in,
                              void* d_out, int out_size, void* d_ws, size_t ws_size,
                              hipStream_t stream) {
  (void)in_sizes; (void)n_in; (void)out_size; (void)ws_size;
  const float* x0 = (const float*)d_in[0];   // (64,576,1024) == (8,4608,1024) view
  const float* w1 = (const float*)d_in[1];   // (1024,4096)
  const float* b1 = (const float*)d_in[2];   // (4096)
  const float* w2 = (const float*)d_in[3];   // (4096,4096)
  const float* b2 = (const float*)d_in[4];   // (4096)

  char* ws = (char*)d_ws;                    // ~171MB total layout
  float*              XA  = (float*)             (ws + 0);          // 75,497,472
  float*              XB  = (float*)             (ws + 75497472);   // 37,748,736
  float*              SZA = (float*)             (ws + 113246208);  // 73,728
  float*              SZB = (float*)             (ws + 113319936);  // 36,864
  _Float16*           MA  = (_Float16*)          (ws + 113356800);  // 2,359,296
  _Float16*           MB  = (_Float16*)          (ws + 115716096);  // 2,359,296
  unsigned long long* NB  = (unsigned long long*)(ws + 118075392);  // 147,456
  int*                NJ  = (int*)               (ws + 118222848);  // 73,728
  int*                EI  = (int*)               (ws + 118296576);  // 73,728
  _Float16*           W1T = (_Float16*)          (ws + 118370304);  // 8,388,608
  _Float16*           W2T = (_Float16*)          (ws + 126758912);  // 33,554,432
  _Float16*           XH  = (_Float16*)          (ws + 160313344);  // 2,097,152
  _Float16*           HH  = (_Float16*)          (ws + 162410496);  // 8,388,608

  static const int Ps[6] = {4608, 2304, 1152, 576, 288, 144};
  static const int Rs[6] = {2304, 1152,  576, 288, 144,  16};

  const float* xin  = x0;
  const float* szin = nullptr;               // size == 1 on first round
  float* xbufs[2]  = {XA, XB};
  float* szbufs[2] = {SZA, SZB};

  for (int itr = 0; itr < 6; ++itr) {
    int p = Ps[itr], r = Rs[itr];
    int HP = p / 2, pnew = p - r, u = HP - r;
    int HPpad = (HP + 15) & ~15;
    int npow2 = 1; while (npow2 < HP) npow2 <<= 1;
    float* xout  = xbufs[itr & 1];
    float* szout = szbufs[itr & 1];

    long items = (long)NBATCH * HPpad * 2;
    metric_kernel<<<(unsigned)((items + 3) / 4), 128, 0, stream>>>(xin, MA, MB, p, HP, HPpad);

    int nbc = NBATCH * S_MAX;
    clear_u64<<<(nbc + 255) / 256, 256, 0, stream>>>(NB, nbc);

    scores_argmax_kernel<<<dim3(HPpad / 16, NBATCH), 128, 0, stream>>>(MA, MB, NB, HP, HPpad);

    sort_kernel<<<NBATCH, 512, 0, stream>>>(NB, NJ, EI, HP, npow2);

    long ncx = (long)NBATCH * pnew * 1024;
    merge_copy_x<<<(unsigned)((ncx + 255) / 256), 256, 0, stream>>>(xin, szin, EI, xout, p, r, u, pnew);
    int ncs = NBATCH * pnew;
    merge_copy_size<<<(ncs + 255) / 256, 256, 0, stream>>>(szin, EI, szout, p, r, u, pnew);
    long nsx = (long)NBATCH * r * 1024;
    merge_scatter_x<<<(unsigned)((nsx + 255) / 256), 256, 0, stream>>>(xin, szin, EI, NJ, xout, p, r, u, pnew);
    int nss = NBATCH * r;
    merge_scatter_size<<<(nss + 255) / 256, 256, 0, stream>>>(szin, EI, NJ, szout, p, r, u, pnew);
    merge_finalize<<<(unsigned)((ncx + 255) / 256), 256, 0, stream>>>(xout, szout, pnew);

    xin = xout; szin = szout;
  }

  // MLP on (8*128=1024, 1024) tokens.
  convert_f16<<<(1024 * 1024 + 255) / 256, 256, 0, stream>>>(xin, XH, 1024L * 1024);
  transpose_f16_tiled<<<dim3(4096 / 32, 1024 / 32), dim3(32, 8), 0, stream>>>(w1, W1T, 1024, 4096);
  transpose_f16_tiled<<<dim3(4096 / 32, 4096 / 32), dim3(32, 8), 0, stream>>>(w2, W2T, 4096, 4096);

  // GEMM1 + bias + exact GELU -> f16, then GEMM2 + bias -> f32 d_out.
  // tiles = (1024/16)*(4096/64) = 4096 waves; 8 waves/block -> 512 blocks.
  gemm_wmma<true ><<<512, 256, 0, stream>>>(XH, W1T, b1, HH, nullptr, 1024, 4096, 1024);
  gemm_wmma<false><<<512, 256, 0, stream>>>(HH, W2T, b2, nullptr, (float*)d_out, 1024, 4096, 4096);
}
